// SymbolicOperator_49744311222435
// MI455X (gfx1250) — compile-verified, matrix operand order checked by reading, not run on hardware
//
#include <hip/hip_runtime.h>
#include <hip/hip_bf16.h>
#include <math.h>

// ---------------------------------------------------------------------------
// Problem constants (from reference)
//   K=128, NP=3, H=384, P=200 (pad 224), L=50 (pad 64), Vin=1000, Vout=512
//   B=256, I=16, STEPS=3  -> 48 global steps
// ---------------------------------------------------------------------------
#define NB       256
#define NI       16
#define NL       50
#define NK       128
#define NH       384
#define NP_PAD   224
#define NVOUT    512
#define GSTEPS   48

typedef __attribute__((ext_vector_type(16))) _Float16 v16h;
typedef __attribute__((ext_vector_type(8)))  float    v8f;

// CDNA5 dense WMMA f32 = f16 x f16 + f32, 16x16x32
__device__ __forceinline__ v8f wmma16(v16h a, v16h b, v8f c) {
  return __builtin_amdgcn_wmma_f32_16x16x32_f16(
      /*neg_a=*/false, a, /*neg_b=*/false, b,
      /*c_mod=*/(short)0, c, /*reuse_a=*/false, /*reuse_b=*/false);
}

// CDNA5 16-bit A-fragment (16x32) element placement:
//   lane = m + 16*((k>>3)&1);  element e = (k&7) + 8*(k>>4)
// (B fragments use the same formula with n in place of m.)
__device__ __forceinline__ int fragIdx(int mn, int k) {
  int lane = mn + (((k >> 3) & 1) << 4);
  int e    = (k & 7) + ((k >> 4) << 3);
  return lane * 16 + e;  // index in halves within a 512-half k-tile block
}

__device__ __forceinline__ float sigmoidf(float x) {
  return 1.0f / (1.0f + expf(-x));
}

// ---------------------------------------------------------------------------
// K0: pre-swizzle weights into WMMA B-fragment layout (f16)
//   wihF : [nt=72][kt=7 ][lane=32][e=16]  from w_ih (1152 x 200), zero pad K
//   whhF : [nt=72][kt=12][lane=32][e=16]  from w_hh (1152 x 384)
//   keysF: [nt=4 ][kt=4 ][lane=32][e=16]  from keys^T (128 x 64), zero pad L
// ---------------------------------------------------------------------------
#define N_WIHF (72 * 7 * 32 * 16)
#define N_WHHF (72 * 12 * 32 * 16)
#define N_KEYF (4 * 4 * 32 * 16)

__global__ void __launch_bounds__(256) prep_kernel(
    const float* __restrict__ w_ih, const float* __restrict__ w_hh,
    const float* __restrict__ keys,
    _Float16* __restrict__ wihF, _Float16* __restrict__ whhF,
    _Float16* __restrict__ keysF) {
  int idx = blockIdx.x * 256 + threadIdx.x;
  if (idx < N_WIHF) {
    int e = idx & 15, lane = (idx >> 4) & 31;
    int kt = (idx >> 9) % 7, nt = (idx >> 9) / 7;
    int nl = lane & 15, khalf = lane >> 4;
    int kl = (e & 7) | (khalf << 3) | ((e >> 3) << 4);
    int kg = kt * 32 + kl, ng = nt * 16 + nl;
    float v = (kg < 200) ? w_ih[ng * 200 + kg] : 0.0f;
    wihF[idx] = (_Float16)v;
  } else if (idx < N_WIHF + N_WHHF) {
    int j = idx - N_WIHF;
    int e = j & 15, lane = (j >> 4) & 31;
    int kt = (j >> 9) % 12, nt = (j >> 9) / 12;
    int nl = lane & 15, khalf = lane >> 4;
    int kl = (e & 7) | (khalf << 3) | ((e >> 3) << 4);
    int kg = kt * 32 + kl, ng = nt * 16 + nl;
    whhF[j] = (_Float16)w_hh[ng * NH + kg];
  } else if (idx < N_WIHF + N_WHHF + N_KEYF) {
    int j = idx - N_WIHF - N_WHHF;
    int e = j & 15, lane = (j >> 4) & 31;
    int kt = (j >> 9) % 4, nt = (j >> 9) / 4;
    int nl = lane & 15, khalf = lane >> 4;
    int kl = (e & 7) | (khalf << 3) | ((e >> 3) << 4);
    int kg = kt * 32 + kl;
    int l = nt * 16 + nl;
    float v = (l < NL) ? keys[l * NK + kg] : 0.0f;
    keysF[j] = (_Float16)v;
  }
}

// ---------------------------------------------------------------------------
// K1: h-trajectory. 16 blocks x 256 threads (8 wave32). Block = 16 batch rows.
//   Per step: scores = h_ptr @ keys^T via WMMA -> softmax -> attn/wm to ws;
//   GRU: gi/gh via WMMA, gate epilogue in C-fragment registers, write h
//   (double-buffered A-fragment layout in LDS).
// ---------------------------------------------------------------------------
__global__ void __launch_bounds__(256) traj_kernel(
    const int* __restrict__ instr, const float* __restrict__ gate_emb,
    const float* __restrict__ program_emb,
    const float* __restrict__ b_ih, const float* __restrict__ b_hh,
    const float* __restrict__ keys,
    const _Float16* __restrict__ wihF, const _Float16* __restrict__ whhF,
    const _Float16* __restrict__ keysF,
    float* __restrict__ attnWS, float* __restrict__ wmWS,
    float* __restrict__ gatesWS) {
  __shared__ __align__(32) _Float16 hA[2][12 * 32 * 16];  // 12 k-tiles, A-layout
  __shared__ __align__(32) _Float16 pA[7 * 32 * 16];      // program, A-layout
  __shared__ float scoresLDS[2][16][64];

  const int tid  = threadIdx.x;
  const int lane = tid & 31;
  const int w    = tid >> 5;  // 8 waves
  const int bbase = blockIdx.x * 16;

  // h0[m][j] = scratch_keys[0][j % 128]
  for (int idx = tid; idx < 16 * NH; idx += 256) {
    int m = idx / NH, j = idx % NH;
    hA[0][(j >> 5) * 512 + fragIdx(m, j & 31)] = (_Float16)keys[j & (NK - 1)];
  }
  int buf = 0;
  __syncthreads();

  for (int i = 0; i < NI; ++i) {
    // Build program A-fragments for this instruction word
    for (int idx = tid; idx < 16 * NP_PAD; idx += 256) {
      int m = idx / NP_PAD, k = idx % NP_PAD;
      int word = instr[i * NB + bbase + m];
      float v = (k < 200) ? program_emb[word * 200 + k] : 0.0f;
      pA[(k >> 5) * 512 + fragIdx(m, k & 31)] = (_Float16)v;
    }
    if (tid < 16) {  // 2-way gate softmax per batch row
      int b = bbase + tid;
      int word = instr[i * NB + b];
      float e0 = gate_emb[word * 2 + 0], e1 = gate_emb[word * 2 + 1];
      float mx = fmaxf(e0, e1);
      float g0 = expf(e0 - mx), g1 = expf(e1 - mx);
      float inv = 1.0f / (g0 + g1);
      gatesWS[(i * NB + b) * 2 + 0] = g0 * inv;
      gatesWS[(i * NB + b) * 2 + 1] = g1 * inv;
    }
    __syncthreads();

    for (int s = 0; s < 3; ++s) {
      const int gstep = i * 3 + s;
      const v16h* hfr = (const v16h*)(&hA[buf][0]);
      const v16h* pfr = (const v16h*)(&pA[0]);

      // ---- scores: waves 0-3 read_ptr (k-tiles 4..7), waves 4-7 write_ptr ----
      {
        int side = w >> 2, nt = w & 3;
        v8f c = {};
#pragma unroll
        for (int kk = 0; kk < 4; ++kk) {
          v16h a = hfr[(4 + side * 4 + kk) * 32 + lane];
          v16h b = *(const v16h*)(keysF + ((nt * 4 + kk) * 32 + lane) * 16);
          c = wmma16(a, b, c);
        }
        int mrow = (lane >> 4) * 8;
        int col = (lane & 15) + nt * 16;
#pragma unroll
        for (int r = 0; r < 8; ++r) scoresLDS[side][r + mrow][col] = c[r];
      }
      __syncthreads();

      // ---- row softmax over L=50, store attn / write-mask to workspace ----
      if (tid < 32) {
        int side = tid >> 4, m = tid & 15;
        float mx = -1e30f;
        for (int l = 0; l < NL; ++l) mx = fmaxf(mx, scoresLDS[side][m][l]);
        float sum = 0.0f;
        for (int l = 0; l < NL; ++l) sum += expf(scoresLDS[side][m][l] - mx);
        float inv = 1.0f / sum;
        float* dst = (side ? wmWS : attnWS) +
                     ((size_t)gstep * NB + bbase + m) * NL;
        for (int l = 0; l < NL; ++l)
          dst[l] = expf(scoresLDS[side][m][l] - mx) * inv;
      }
      __syncthreads();

      // ---- GRU: each wave owns 3 triplets {t, t+24, t+48} of N-tiles ----
      for (int j = 0; j < 3; ++j) {
        int t = w * 3 + j;
        v8f cir = {}, ciz = {}, cin = {}, chr_ = {}, chz = {}, chn = {};
#pragma unroll
        for (int kt = 0; kt < 7; ++kt) {
          v16h a = pfr[kt * 32 + lane];
          v16h b0 = *(const v16h*)(wihF + (((t)      * 7 + kt) * 32 + lane) * 16);
          v16h b1 = *(const v16h*)(wihF + (((t + 24) * 7 + kt) * 32 + lane) * 16);
          v16h b2 = *(const v16h*)(wihF + (((t + 48) * 7 + kt) * 32 + lane) * 16);
          cir = wmma16(a, b0, cir);
          ciz = wmma16(a, b1, ciz);
          cin = wmma16(a, b2, cin);
        }
#pragma unroll
        for (int kt = 0; kt < 12; ++kt) {
          v16h a = hfr[kt * 32 + lane];
          v16h b0 = *(const v16h*)(whhF + (((t)      * 12 + kt) * 32 + lane) * 16);
          v16h b1 = *(const v16h*)(whhF + (((t + 24) * 12 + kt) * 32 + lane) * 16);
          v16h b2 = *(const v16h*)(whhF + (((t + 48) * 12 + kt) * 32 + lane) * 16);
          chr_ = wmma16(a, b0, chr_);
          chz  = wmma16(a, b1, chz);
          chn  = wmma16(a, b2, chn);
        }
        // gate epilogue in C-layout registers
        int n = lane & 15;
        int jj = t * 16 + n;
        float bir = b_ih[jj],       bhr = b_hh[jj];
        float biz = b_ih[NH + jj],  bhz = b_hh[NH + jj];
        float bin = b_ih[2*NH + jj], bhn = b_hh[2*NH + jj];
        int mrow = (lane >> 4) * 8;
        int kt2 = jj >> 5, kl2 = jj & 31;
#pragma unroll
        for (int r = 0; r < 8; ++r) {
          int m = r + mrow;
          float rg = sigmoidf(cir[r] + bir + chr_[r] + bhr);
          float zg = sigmoidf(ciz[r] + biz + chz[r] + bhz);
          float ng = tanhf(cin[r] + bin + rg * (chn[r] + bhn));
          int fi = kt2 * 512 + fragIdx(m, kl2);
          float hold = (float)hA[buf][fi];
          hA[buf ^ 1][fi] = (_Float16)((1.0f - zg) * ng + zg * hold);
        }
      }
      buf ^= 1;
      __syncthreads();
    }
  }
}

// ---------------------------------------------------------------------------
// K2: vals evolution + log-softmax. 256 blocks (one per b) x 512 threads
// (one per v). 50-element vals column lives in registers for all 48 steps.
// ---------------------------------------------------------------------------
__global__ void __launch_bounds__(512) vals_kernel(
    const int* __restrict__ instr, const int* __restrict__ true_actions,
    const float* __restrict__ primitive_emb, const float* __restrict__ init_value,
    const float* __restrict__ attnWS, const float* __restrict__ wmWS,
    const float* __restrict__ gatesWS, float* __restrict__ out) {
  __shared__ float attn_s[NL];
  __shared__ float wm_s[NL];
  __shared__ float red[17];

  const int b = blockIdx.x;
  const int tid = threadIdx.x;
  const int v = tid;
  const int lane = tid & 31, wid = tid >> 5;

  float vals[NL];
  float iv = init_value[v];
#pragma unroll
  for (int l = 0; l < NL; ++l) vals[l] = iv;

  float g0 = 0.0f, g1 = 0.0f, prim = 0.0f;
  for (int gs = 0; gs < GSTEPS; ++gs) {
    if (gs % 3 == 0) {
      int i = gs / 3;
      int word = instr[i * NB + b];
      prim = primitive_emb[(size_t)word * NVOUT + v];
      g0 = gatesWS[(i * NB + b) * 2 + 0];
      g1 = gatesWS[(i * NB + b) * 2 + 1];
    }
    if (tid < NL) {
      attn_s[tid] = attnWS[((size_t)gs * NB + b) * NL + tid];
      wm_s[tid]   = wmWS[((size_t)gs * NB + b) * NL + tid];
    }
    __syncthreads();
    float rv = 0.0f;
#pragma unroll
    for (int l = 0; l < NL; ++l) rv += attn_s[l] * vals[l];
    float nv = g0 * prim + g1 * rv;
#pragma unroll
    for (int l = 0; l < NL; ++l) {
      float wm = wm_s[l];
      vals[l] = wm * nv + (1.0f - wm) * vals[l];
    }
    __syncthreads();
  }

  // log_softmax over v (512 threads) per l; write actions[b][v][l]
  for (int l = 0; l < NL; ++l) {
    float x = vals[l];
    float mx = x;
#pragma unroll
    for (int o = 16; o > 0; o >>= 1) mx = fmaxf(mx, __shfl_xor(mx, o));
    if (lane == 0) red[wid] = mx;
    __syncthreads();
    if (tid == 0) {
      float m2 = red[0];
      for (int k = 1; k < 16; ++k) m2 = fmaxf(m2, red[k]);
      red[16] = m2;
    }
    __syncthreads();
    float mAll = red[16];
    float ex = expf(x - mAll);
    float sm = ex;
#pragma unroll
    for (int o = 16; o > 0; o >>= 1) sm += __shfl_xor(sm, o);
    __syncthreads();  // red[0..15] free for reuse
    if (lane == 0) red[wid] = sm;
    __syncthreads();
    if (tid == 0) {
      float s2 = 0.0f;
      for (int k = 0; k < 16; ++k) s2 += red[k];
      red[16] = s2;
    }
    __syncthreads();
    float lse = logf(red[16]);
    out[((size_t)b * NVOUT + v) * NL + l] = x - mAll - lse;
    __syncthreads();
  }
  // true_actions.T as second output (numeric cast to float)
  if (tid < NL) {
    out[(size_t)NB * NVOUT * NL + (size_t)b * NL + tid] =
        (float)true_actions[tid * NB + b];
  }
}

// ---------------------------------------------------------------------------
// Launch
// ---------------------------------------------------------------------------
extern "C" void kernel_launch(void* const* d_in, const int* in_sizes, int n_in,
                              void* d_out, int out_size, void* d_ws,
                              size_t ws_size, hipStream_t stream) {
  const int*   instr    = (const int*)d_in[0];
  const int*   tact     = (const int*)d_in[1];
  const float* gate_emb = (const float*)d_in[2];
  const float* prog_emb = (const float*)d_in[3];
  const float* prim_emb = (const float*)d_in[4];
  const float* w_ih     = (const float*)d_in[5];
  const float* w_hh     = (const float*)d_in[6];
  const float* b_ih     = (const float*)d_in[7];
  const float* b_hh     = (const float*)d_in[8];
  const float* keys     = (const float*)d_in[9];
  const float* initv    = (const float*)d_in[10];

  char* ws = (char*)d_ws;
  size_t off = 0;
  _Float16* wihF  = (_Float16*)(ws + off); off += (size_t)N_WIHF * 2;  // 516096
  _Float16* whhF  = (_Float16*)(ws + off); off += (size_t)N_WHHF * 2;  // 884736
  _Float16* keysF = (_Float16*)(ws + off); off += (size_t)N_KEYF * 2;  // 16384
  float* attnWS   = (float*)(ws + off);    off += (size_t)GSTEPS * NB * NL * 4;
  float* wmWS     = (float*)(ws + off);    off += (size_t)GSTEPS * NB * NL * 4;
  float* gatesWS  = (float*)(ws + off);    off += (size_t)NI * NB * 2 * 4;
  // total ~6.4 MB of d_ws

  const int totalPrep = N_WIHF + N_WHHF + N_KEYF;  // 708608 -> 2768 blocks
  prep_kernel<<<(totalPrep + 255) / 256, 256, 0, stream>>>(
      w_ih, w_hh, keys, wihF, whhF, keysF);

  traj_kernel<<<NB / 16, 256, 0, stream>>>(
      instr, gate_emb, prog_emb, b_ih, b_hh, keys, wihF, whhF, keysF,
      attnWS, wmWS, gatesWS);

  vals_kernel<<<NB, 512, 0, stream>>>(
      instr, tact, prim_emb, initv, attnWS, wmWS, gatesWS, (float*)d_out);
}
